// SimpleNeuroSAT_40767829574340
// MI455X (gfx1250) — compile-verified, matrix operand order checked by reading, not run on hardware
//
#include <hip/hip_runtime.h>

#define NVARS    50000
#define NLITS    100000
#define NCLAUSES 210000
#define FDIM     80

typedef __attribute__((ext_vector_type(16))) __bf16 v16bf;
typedef __attribute__((ext_vector_type(8)))  float  v8f;

// ---------------------------------------------------------------------------
// CDNA5 16-bit WMMA fragment layout helpers (wave32).
// A fragment (16x32 bf16): lane = m + 16*h holds, in element e = 2j+b,
//   k_local = ((j<4) ? 2j : 8+2j) + 8h + b.
// We store tiles in "fragment order": for each k-step a 512-element record
// [32 lanes][16 elems], so each lane's fragment is one contiguous 32B vector.
// B fragments use the same mapping with m -> n.
// ---------------------------------------------------------------------------
__device__ __forceinline__ int frag_slot(int m, int kl) {
    int h, j;
    if (kl < 16) { h = (kl >> 3) & 1; j = (kl & 7) >> 1; }
    else         { int k2 = kl - 16; h = (k2 >> 3) & 1; j = 4 + ((k2 & 7) >> 1); }
    int b = kl & 1;
    return (m + 16 * h) * 16 + (2 * j + b);
}

// Load A fragment from fragment-ordered LDS tile (one 32B read per lane).
__device__ __forceinline__ v16bf load_af(const __bf16* tile, int ks) {
    const v16bf* p = (const v16bf*)tile;
    return p[ks * 32 + (threadIdx.x & 31)];
}

// Load B fragment from fragment-ordered global weights.
__device__ __forceinline__ v16bf load_bf(const __bf16* wf, int ntiles, int ks, int t) {
    const v16bf* p = (const v16bf*)wf;
    return p[(ks * ntiles + t) * 32 + (threadIdx.x & 31)];
}

__device__ __forceinline__ float relu6f(float x) {
    return fminf(fmaxf(x, 0.0f), 6.0f);
}

// ---------------------------------------------------------------------------
// Utility kernels
// ---------------------------------------------------------------------------
__global__ void sns_fill_kernel(float* __restrict__ p, long n, const float* vptr) {
    float v = vptr ? vptr[0] : 0.0f;
    long i = (long)blockIdx.x * blockDim.x + threadIdx.x;
    long stride = (long)gridDim.x * blockDim.x;
    for (; i < n; i += stride) p[i] = v;
}

// fp32 weights [K][N] -> bf16 fragment order [Ksteps][Ntiles][32 lanes][16 elems],
// zero-filled outside [K][N].
__global__ void sns_cvtfrag_kernel(const float* __restrict__ src, __bf16* __restrict__ dst,
                                   int K, int N, int Ksteps, int Ntiles) {
    long total = (long)Ksteps * Ntiles * 512;
    long i = (long)blockIdx.x * blockDim.x + threadIdx.x;
    long stride = (long)gridDim.x * blockDim.x;
    for (; i < total; i += stride) {
        int e    = (int)(i & 15);
        int lane = (int)((i >> 4) & 31);
        int ft   = (int)(i >> 9);
        int t  = ft % Ntiles;
        int ks = ft / Ntiles;
        int half = lane >> 4, n = lane & 15;
        int j = e >> 1, b = e & 1;
        int kl = ((j < 4) ? 2 * j : 8 + 2 * j) + 8 * half + b;
        int k = ks * 32 + kl;
        int nc = t * 16 + n;
        float v = (k < K && nc < N) ? src[(long)k * N + nc] : 0.0f;
        dst[i] = (__bf16)v;
    }
}

// dst[didx[e]][f] += src[sidx[e]][f]
__global__ void sns_scatter_kernel(const float* __restrict__ src, const int* __restrict__ sidx,
                                   float* dst, const int* __restrict__ didx, int nnz) {
    long total = (long)nnz * FDIM;
    long i = (long)blockIdx.x * blockDim.x + threadIdx.x;
    long stride = (long)gridDim.x * blockDim.x;
    for (; i < total; i += stride) {
        int e = (int)(i / FDIM);
        int f = (int)(i % FDIM);
        atomicAdd(&dst[(long)didx[e] * FDIM + f], src[(long)sidx[e] * FDIM + f]);
    }
}

// Column sums / sumsq over [nrows][80] -> sums[0..79]=sum, sums[80..159]=sumsq
__global__ void sns_colstats_kernel(const float* __restrict__ x, int nrows,
                                    float* __restrict__ sums) {
    __shared__ float s_sum[FDIM];
    __shared__ float s_sq[FDIM];
    int tid = threadIdx.x;
    if (tid < FDIM) { s_sum[tid] = 0.0f; s_sq[tid] = 0.0f; }
    __syncthreads();
    long total = (long)nrows * FDIM;
    long i = (long)blockIdx.x * blockDim.x + tid;
    long stride = (long)gridDim.x * blockDim.x;
    for (; i < total; i += stride) {
        float v = x[i];
        int c = (int)(i % FDIM);
        atomicAdd(&s_sum[c], v);
        atomicAdd(&s_sq[c], v * v);
    }
    __syncthreads();
    if (tid < FDIM) {
        atomicAdd(&sums[tid], s_sum[tid]);
        atomicAdd(&sums[FDIM + tid], s_sq[tid]);
    }
}

__global__ void sns_normalize_kernel(const float* __restrict__ in, float* __restrict__ outp,
                                     const float* __restrict__ sums, int nrows) {
    long total = (long)nrows * FDIM;
    float inv_n = 1.0f / (float)nrows;
    long i = (long)blockIdx.x * blockDim.x + threadIdx.x;
    long stride = (long)gridDim.x * blockDim.x;
    for (; i < total; i += stride) {
        int c = (int)(i % FDIM);
        float m   = sums[c] * inv_n;
        float var = sums[FDIM + c] * inv_n - m * m;
        outp[i] = (in[i] - m) * rsqrtf(var + 1e-3f);
    }
}

// ---------------------------------------------------------------------------
// C update: relu6([C | LC*s] @ W1 + b1) @ W2 + b2    (160 -> 160 -> 80)
// out may alias LC (each block reads its rows before writing them).
// ---------------------------------------------------------------------------
__global__ __launch_bounds__(128) void sns_c_mlp_kernel(
    const float* __restrict__ Cin, const float* LCm, const float* __restrict__ lc_scale,
    const __bf16* __restrict__ W1f, const float* __restrict__ b1,
    const __bf16* __restrict__ W2f, const float* __restrict__ b2,
    float* out, int nrows) {
    __shared__ __align__(32) __bf16 s_in[4 * 5 * 512];  // 4 tiles, 5 k-steps
    __shared__ __align__(32) __bf16 s_h[4 * 5 * 512];
    int tid = threadIdx.x, wave = tid >> 5, lane = tid & 31;
    int half = lane >> 4, mn = lane & 15;
    int row0 = blockIdx.x * 64;
    float scale = lc_scale[0];

    // stage [64][160] input into A-fragment order
    for (int i = tid; i < 64 * 160; i += 128) {
        int r = i / 160, c = i % 160;
        int gr = row0 + r;
        float v = 0.0f;
        if (gr < nrows)
            v = (c < 80) ? Cin[(long)gr * FDIM + c] : LCm[(long)gr * FDIM + (c - 80)] * scale;
        s_in[(r >> 4) * (5 * 512) + (c >> 5) * 512 + frag_slot(r & 15, c & 31)] = (__bf16)v;
    }
    __syncthreads();

    const __bf16* a_tile = &s_in[wave * (5 * 512)];
    v8f acc[10];
#pragma unroll
    for (int t = 0; t < 10; ++t) acc[t] = (v8f){0, 0, 0, 0, 0, 0, 0, 0};
    for (int ks = 0; ks < 5; ++ks) {
        v16bf a = load_af(a_tile, ks);
#pragma unroll
        for (int t = 0; t < 10; ++t) {
            v16bf b = load_bf(W1f, 10, ks, t);
            acc[t] = __builtin_amdgcn_wmma_f32_16x16x32_bf16(false, a, false, b,
                                                             (short)0, acc[t], false, false);
        }
    }
    // relu6(acc + b1) -> h tile (fragment order; col is K of GEMM2)
    __bf16* h_tile = &s_h[wave * (5 * 512)];
#pragma unroll
    for (int t = 0; t < 10; ++t) {
        int col = t * 16 + mn;
        float bias = b1[col];
        int base = (col >> 5) * 512;
        int kl = col & 31;
#pragma unroll
        for (int v = 0; v < 8; ++v)
            h_tile[base + frag_slot(v + 8 * half, kl)] = (__bf16)relu6f(acc[t][v] + bias);
    }
    __syncthreads();

    v8f acc2[5];
#pragma unroll
    for (int t = 0; t < 5; ++t) acc2[t] = (v8f){0, 0, 0, 0, 0, 0, 0, 0};
    for (int ks = 0; ks < 5; ++ks) {
        v16bf a = load_af(h_tile, ks);
#pragma unroll
        for (int t = 0; t < 5; ++t) {
            v16bf b = load_bf(W2f, 5, ks, t);
            acc2[t] = __builtin_amdgcn_wmma_f32_16x16x32_bf16(false, a, false, b,
                                                              (short)0, acc2[t], false, false);
        }
    }
#pragma unroll
    for (int t = 0; t < 5; ++t) {
        int col = t * 16 + mn;
        float bias = b2[col];
#pragma unroll
        for (int v = 0; v < 8; ++v) {
            int gr = row0 + wave * 16 + v + 8 * half;
            if (gr < nrows) out[(long)gr * FDIM + col] = acc2[t][v] + bias;
        }
    }
}

// ---------------------------------------------------------------------------
// L update: relu6([L | CL*s | L_flip] @ W1 + b1) @ W2 + b2  (240 -> 240 -> 80)
// K padded 240 -> 256 (8 k-steps); weights pre-padded in fragment order.
// out may alias CL.
// ---------------------------------------------------------------------------
__global__ __launch_bounds__(128) void sns_l_mlp_kernel(
    const float* __restrict__ Lin, const float* CLm, const float* __restrict__ cl_scale,
    const __bf16* __restrict__ W1f, const float* __restrict__ b1,
    const __bf16* __restrict__ W2f, const float* __restrict__ b2,
    float* out, int nrows, int nvars) {
    __shared__ __align__(32) __bf16 s_in[4 * 8 * 512];  // 4 tiles, 8 k-steps
    __shared__ __align__(32) __bf16 s_h[4 * 8 * 512];
    int tid = threadIdx.x, wave = tid >> 5, lane = tid & 31;
    int half = lane >> 4, mn = lane & 15;
    int row0 = blockIdx.x * 64;
    float scale = cl_scale[0];

    // stage [64][256] (padded) input into A-fragment order
    for (int i = tid; i < 64 * 256; i += 128) {
        int r = i >> 8, c = i & 255;
        int gr = row0 + r;
        float v = 0.0f;
        if (gr < nrows && c < 240) {
            if (c < 80)       v = Lin[(long)gr * FDIM + c];
            else if (c < 160) v = CLm[(long)gr * FDIM + (c - 80)] * scale;
            else {
                int fr = (gr < nvars) ? gr + nvars : gr - nvars;
                v = Lin[(long)fr * FDIM + (c - 160)];
            }
        }
        s_in[(r >> 4) * (8 * 512) + (c >> 5) * 512 + frag_slot(r & 15, c & 31)] = (__bf16)v;
    }
    __syncthreads();

    const __bf16* a_tile = &s_in[wave * (8 * 512)];
    v8f acc[15];
#pragma unroll
    for (int t = 0; t < 15; ++t) acc[t] = (v8f){0, 0, 0, 0, 0, 0, 0, 0};
    for (int ks = 0; ks < 8; ++ks) {
        v16bf a = load_af(a_tile, ks);
#pragma unroll
        for (int t = 0; t < 15; ++t) {
            v16bf b = load_bf(W1f, 15, ks, t);
            acc[t] = __builtin_amdgcn_wmma_f32_16x16x32_bf16(false, a, false, b,
                                                             (short)0, acc[t], false, false);
        }
    }
    // h tile (fragment order), zero the 240..255 K-tail of the last fragment
    __bf16* h_tile = &s_h[wave * (8 * 512)];
    for (int idx = lane; idx < 256; idx += 32) {
        int m = idx >> 4, c = 240 + (idx & 15);
        h_tile[(c >> 5) * 512 + frag_slot(m, c & 31)] = (__bf16)0.0f;
    }
#pragma unroll
    for (int t = 0; t < 15; ++t) {
        int col = t * 16 + mn;
        float bias = b1[col];
        int base = (col >> 5) * 512;
        int kl = col & 31;
#pragma unroll
        for (int v = 0; v < 8; ++v)
            h_tile[base + frag_slot(v + 8 * half, kl)] = (__bf16)relu6f(acc[t][v] + bias);
    }
    __syncthreads();

    v8f acc2[5];
#pragma unroll
    for (int t = 0; t < 5; ++t) acc2[t] = (v8f){0, 0, 0, 0, 0, 0, 0, 0};
    for (int ks = 0; ks < 8; ++ks) {
        v16bf a = load_af(h_tile, ks);
#pragma unroll
        for (int t = 0; t < 5; ++t) {
            v16bf b = load_bf(W2f, 5, ks, t);
            acc2[t] = __builtin_amdgcn_wmma_f32_16x16x32_bf16(false, a, false, b,
                                                              (short)0, acc2[t], false, false);
        }
    }
#pragma unroll
    for (int t = 0; t < 5; ++t) {
        int col = t * 16 + mn;
        float bias = b2[col];
#pragma unroll
        for (int v = 0; v < 8; ++v) {
            int gr = row0 + wave * 16 + v + 8 * half;
            if (gr < nrows) out[(long)gr * FDIM + col] = acc2[t][v] + bias;
        }
    }
}

// ---------------------------------------------------------------------------
// One 160x160 layer of the V MLP (fragment-ordered LDS in/out).
// ---------------------------------------------------------------------------
__device__ __forceinline__ void sns_v_layer(const __bf16* a_base, __bf16* h_base,
                                            const __bf16* __restrict__ Wf,
                                            const float* __restrict__ bias,
                                            int wave, int half, int mn) {
    const __bf16* a_tile = a_base + wave * (5 * 512);
    __bf16* h_tile = h_base + wave * (5 * 512);
    v8f acc[10];
#pragma unroll
    for (int t = 0; t < 10; ++t) acc[t] = (v8f){0, 0, 0, 0, 0, 0, 0, 0};
    for (int ks = 0; ks < 5; ++ks) {
        v16bf a = load_af(a_tile, ks);
#pragma unroll
        for (int t = 0; t < 10; ++t) {
            v16bf b = load_bf(Wf, 10, ks, t);
            acc[t] = __builtin_amdgcn_wmma_f32_16x16x32_bf16(false, a, false, b,
                                                             (short)0, acc[t], false, false);
        }
    }
#pragma unroll
    for (int t = 0; t < 10; ++t) {
        int col = t * 16 + mn;
        float bb = bias[col];
        int base = (col >> 5) * 512;
        int kl = col & 31;
#pragma unroll
        for (int v = 0; v < 8; ++v)
            h_tile[base + frag_slot(v + 8 * half, kl)] = (__bf16)relu6f(acc[t][v] + bb);
    }
    __syncthreads();
}

// ---------------------------------------------------------------------------
// V score MLP: 3x (160x160 relu6) then dot with Vw4 (160x1).
// Input row i = [L[i], L[i+nvars]].
// ---------------------------------------------------------------------------
__global__ __launch_bounds__(128) void sns_v_mlp_kernel(
    const float* __restrict__ Lin,
    const __bf16* __restrict__ W1f, const float* __restrict__ b1,
    const __bf16* __restrict__ W2f, const float* __restrict__ b2,
    const __bf16* __restrict__ W3f, const float* __restrict__ b3,
    const float* __restrict__ W4, const float* __restrict__ b4,
    float* __restrict__ scores, int nvars) {
    __shared__ __align__(32) __bf16 s_a[4 * 5 * 512];
    __shared__ __align__(32) __bf16 s_b[4 * 5 * 512];
    int tid = threadIdx.x, wave = tid >> 5, lane = tid & 31;
    int half = lane >> 4, mn = lane & 15;
    int row0 = blockIdx.x * 64;

    for (int i = tid; i < 64 * 160; i += 128) {
        int r = i / 160, c = i % 160;
        int gr = row0 + r;
        float v = 0.0f;
        if (gr < nvars)
            v = (c < 80) ? Lin[(long)gr * FDIM + c] : Lin[(long)(gr + nvars) * FDIM + (c - 80)];
        s_a[(r >> 4) * (5 * 512) + (c >> 5) * 512 + frag_slot(r & 15, c & 31)] = (__bf16)v;
    }
    __syncthreads();

    sns_v_layer(s_a, s_b, W1f, b1, wave, half, mn);  // h1 -> s_b
    sns_v_layer(s_b, s_a, W2f, b2, wave, half, mn);  // h2 -> s_a
    sns_v_layer(s_a, s_b, W3f, b3, wave, half, mn);  // h3 -> s_b

    // Final 160 -> 1 dot (h3 in s_b, fragment order); one thread per row.
    if (tid < 64) {
        int gr = row0 + tid;
        if (gr < nvars) {
            const __bf16* tile = &s_b[(tid >> 4) * (5 * 512)];
            int m = tid & 15;
            float s = 0.0f;
#pragma unroll 4
            for (int k = 0; k < 160; ++k)
                s += (float)tile[(k >> 5) * 512 + frag_slot(m, k & 31)] * W4[k];
            scores[gr] = s + b4[0];
        }
    }
}

// ---------------------------------------------------------------------------
// Loss
// ---------------------------------------------------------------------------
__global__ void sns_loss_edges_kernel(const int* __restrict__ edge_lit,
                                      const int* __restrict__ edge_clause,
                                      const float* __restrict__ scores,
                                      float* __restrict__ clause_sum, int nnz) {
    int i = blockIdx.x * blockDim.x + threadIdx.x;
    if (i >= nnz) return;
    int lit = edge_lit[i];
    float sgn = (lit < NVARS) ? 1.0f : -1.0f;
    int var = (lit < NVARS) ? lit : lit - NVARS;
    float x = scores[var] * sgn;
    float sp = (x > 20.0f) ? x : log1pf(expf(x));
    atomicAdd(&clause_sum[edge_clause[i]], sp);
}

__global__ void sns_loss_clauses_kernel(const float* __restrict__ clause_sum,
                                        float* __restrict__ loss, int n) {
    __shared__ float red[256];
    int tid = threadIdx.x;
    float local = 0.0f;
    for (long i = (long)blockIdx.x * blockDim.x + tid; i < n;
         i += (long)gridDim.x * blockDim.x) {
        float cv = expf(-clause_sum[i]);
        local += cv * (-logf(1.0f - cv + 1e-8f));
    }
    red[tid] = local;
    __syncthreads();
    for (int s = 128; s > 0; s >>= 1) {
        if (tid < s) red[tid] += red[tid + s];
        __syncthreads();
    }
    if (tid == 0) atomicAdd(loss, red[0]);
}

// ---------------------------------------------------------------------------
// Host launcher
// ---------------------------------------------------------------------------
static inline int gridFor(long n, int block) { return (int)((n + block - 1) / block); }

extern "C" void kernel_launch(void* const* d_in, const int* in_sizes, int n_in,
                              void* d_out, int out_size, void* d_ws, size_t ws_size,
                              hipStream_t stream) {
    const int*   edge_lit    = (const int*)d_in[0];
    const int*   edge_clause = (const int*)d_in[1];
    const float* L_init      = (const float*)d_in[4];
    const float* C_init      = (const float*)d_in[5];
    const float* LC_scale    = (const float*)d_in[6];
    const float* CL_scale    = (const float*)d_in[7];
    const float* Cw1 = (const float*)d_in[8],  *Cb1 = (const float*)d_in[9];
    const float* Cw2 = (const float*)d_in[10], *Cb2 = (const float*)d_in[11];
    const float* Lw1 = (const float*)d_in[12], *Lb1 = (const float*)d_in[13];
    const float* Lw2 = (const float*)d_in[14], *Lb2 = (const float*)d_in[15];
    const float* Vw1 = (const float*)d_in[16], *Vb1 = (const float*)d_in[17];
    const float* Vw2 = (const float*)d_in[18], *Vb2 = (const float*)d_in[19];
    const float* Vw3 = (const float*)d_in[20], *Vb3 = (const float*)d_in[21];
    const float* Vw4 = (const float*)d_in[22], *Vb4 = (const float*)d_in[23];
    const int nnz = in_sizes[0];

    // workspace carve
    char* p = (char*)d_ws;
    auto carve = [&](size_t bytes) -> char* {
        char* r = p;
        p += (bytes + 255) & ~(size_t)255;
        return r;
    };
    float*  L     = (float*)carve((size_t)NLITS * FDIM * 4);
    float*  C     = (float*)carve((size_t)NCLAUSES * FDIM * 4);
    float*  LCbuf = (float*)carve((size_t)NCLAUSES * FDIM * 4);  // LC msgs, aliases Craw
    float*  CLbuf = (float*)carve((size_t)NLITS * FDIM * 4);     // CL msgs, aliases Lraw
    float*  csum  = (float*)carve((size_t)NCLAUSES * 4);
    float*  stats = (float*)carve(160 * 4);
    // fragment-ordered bf16 weights: [Ksteps][Ntiles][32][16]
    __bf16* wC1 = (__bf16*)carve((size_t)5 * 10 * 512 * 2);
    __bf16* wC2 = (__bf16*)carve((size_t)5 * 5  * 512 * 2);
    __bf16* wL1 = (__bf16*)carve((size_t)8 * 15 * 512 * 2);
    __bf16* wL2 = (__bf16*)carve((size_t)8 * 5  * 512 * 2);
    __bf16* wV1 = (__bf16*)carve((size_t)5 * 10 * 512 * 2);
    __bf16* wV2 = (__bf16*)carve((size_t)5 * 10 * 512 * 2);
    __bf16* wV3 = (__bf16*)carve((size_t)5 * 10 * 512 * 2);

    float* scores = (float*)d_out;
    float* loss   = scores + NVARS;

    // weight shuffles (bf16, fragment order, zero-padded)
    sns_cvtfrag_kernel<<<gridFor(5 * 10 * 512, 256), 256, 0, stream>>>(Cw1, wC1, 160, 160, 5, 10);
    sns_cvtfrag_kernel<<<gridFor(5 * 5  * 512, 256), 256, 0, stream>>>(Cw2, wC2, 160, 80, 5, 5);
    sns_cvtfrag_kernel<<<gridFor(8 * 15 * 512, 256), 256, 0, stream>>>(Lw1, wL1, 240, 240, 8, 15);
    sns_cvtfrag_kernel<<<gridFor(8 * 5  * 512, 256), 256, 0, stream>>>(Lw2, wL2, 240, 80, 8, 5);
    sns_cvtfrag_kernel<<<gridFor(5 * 10 * 512, 256), 256, 0, stream>>>(Vw1, wV1, 160, 160, 5, 10);
    sns_cvtfrag_kernel<<<gridFor(5 * 10 * 512, 256), 256, 0, stream>>>(Vw2, wV2, 160, 160, 5, 10);
    sns_cvtfrag_kernel<<<gridFor(5 * 10 * 512, 256), 256, 0, stream>>>(Vw3, wV3, 160, 160, 5, 10);

    // init embeddings
    sns_fill_kernel<<<gridFor((long)NLITS * FDIM, 256), 256, 0, stream>>>(L, (long)NLITS * FDIM, L_init);
    sns_fill_kernel<<<gridFor((long)NCLAUSES * FDIM, 256), 256, 0, stream>>>(C, (long)NCLAUSES * FDIM, C_init);

    const int cBlocks = (NCLAUSES + 63) / 64;
    const int lBlocks = (NLITS + 63) / 64;
    const int vBlocks = (NVARS + 63) / 64;

    for (int round = 0; round < 4; ++round) {
        // LC messages
        sns_fill_kernel<<<gridFor((long)NCLAUSES * FDIM, 256), 256, 0, stream>>>(
            LCbuf, (long)NCLAUSES * FDIM, (const float*)nullptr);
        sns_scatter_kernel<<<gridFor((long)nnz * FDIM, 256), 256, 0, stream>>>(
            L, edge_lit, LCbuf, edge_clause, nnz);
        // C MLP (writes Craw in-place over LCbuf)
        sns_c_mlp_kernel<<<cBlocks, 128, 0, stream>>>(C, LCbuf, LC_scale, wC1, Cb1, wC2, Cb2,
                                                      LCbuf, NCLAUSES);
        // normalize C
        sns_fill_kernel<<<1, 256, 0, stream>>>(stats, 160, (const float*)nullptr);
        sns_colstats_kernel<<<512, 256, 0, stream>>>(LCbuf, NCLAUSES, stats);
        sns_normalize_kernel<<<gridFor((long)NCLAUSES * FDIM, 256), 256, 0, stream>>>(
            LCbuf, C, stats, NCLAUSES);

        // CL messages
        sns_fill_kernel<<<gridFor((long)NLITS * FDIM, 256), 256, 0, stream>>>(
            CLbuf, (long)NLITS * FDIM, (const float*)nullptr);
        sns_scatter_kernel<<<gridFor((long)nnz * FDIM, 256), 256, 0, stream>>>(
            C, edge_clause, CLbuf, edge_lit, nnz);
        // L MLP (writes Lraw in-place over CLbuf)
        sns_l_mlp_kernel<<<lBlocks, 128, 0, stream>>>(L, CLbuf, CL_scale, wL1, Lb1, wL2, Lb2,
                                                      CLbuf, NLITS, NVARS);
        // normalize L
        sns_fill_kernel<<<1, 256, 0, stream>>>(stats, 160, (const float*)nullptr);
        sns_colstats_kernel<<<512, 256, 0, stream>>>(CLbuf, NLITS, stats);
        sns_normalize_kernel<<<gridFor((long)NLITS * FDIM, 256), 256, 0, stream>>>(
            CLbuf, L, stats, NLITS);
    }

    // V scores -> d_out[0..NVARS)
    sns_v_mlp_kernel<<<vBlocks, 128, 0, stream>>>(L, wV1, Vb1, wV2, Vb2, wV3, Vb3,
                                                  Vw4, Vb4, scores, NVARS);

    // loss -> d_out[NVARS]
    sns_fill_kernel<<<gridFor(NCLAUSES, 256), 256, 0, stream>>>(csum, NCLAUSES, (const float*)nullptr);
    sns_fill_kernel<<<1, 32, 0, stream>>>(loss, 1, (const float*)nullptr);
    sns_loss_edges_kernel<<<gridFor(nnz, 256), 256, 0, stream>>>(edge_lit, edge_clause,
                                                                 scores, csum, nnz);
    sns_loss_clauses_kernel<<<256, 256, 0, stream>>>(csum, loss, NCLAUSES);
}
